// SelfAttention_14637248545046
// MI455X (gfx1250) — matrix operand order repaired and checked by reading, not compiled, not run on hardware
//
#include <hip/hip_runtime.h>
#include <hip/hip_bf16.h>
#include <stdint.h>

#define D_   2048
#define S_   2048
#define H_   16
#define DH_  128
#define B_   2
#define M_   (B_ * S_)   // 4096 rows
#define BH_  (B_ * H_)   // 32

typedef __attribute__((ext_vector_type(16))) __bf16          v16bf;
typedef __attribute__((ext_vector_type(8)))  float           v8f;
typedef __attribute__((ext_vector_type(16))) unsigned short  v16us;
typedef __attribute__((ext_vector_type(8)))  unsigned short  v8us;
typedef __attribute__((ext_vector_type(4)))  unsigned short  v4us;

typedef __attribute__((address_space(3))) unsigned short lds_us;

// native bf16 convert (v_cvt_*_bf16_f32 on gfx1250)
__device__ __forceinline__ unsigned short f2bf(float f) {
  __bf16 b = (__bf16)f;
  return __builtin_bit_cast(unsigned short, b);
}

__device__ __forceinline__ v8f wmma_bf16(v16us a, v16us b, v8f c) {
  return __builtin_amdgcn_wmma_f32_16x16x32_bf16(
      false, __builtin_bit_cast(v16bf, a),
      false, __builtin_bit_cast(v16bf, b),
      (short)0, c, false, false);
}

// async DMA: 16 bytes global -> LDS, tracked by ASYNCcnt
__device__ __forceinline__ void async_ld_b128(unsigned lds_off, const void* g) {
  asm volatile("global_load_async_to_lds_b128 %0, %1, off"
               :: "v"(lds_off), "v"((unsigned long long)(uintptr_t)g)
               : "memory");
}
__device__ __forceinline__ void wait_async0() {
  asm volatile("s_wait_asynccnt 0" ::: "memory");
}

// ---------------------------------------------------------------- cast kernel
__global__ void cast_f32_bf16(const float* __restrict__ in,
                              unsigned short* __restrict__ out, int n) {
  int i = (blockIdx.x * blockDim.x + threadIdx.x) * 4;
  if (i + 3 < n) {
    float4 vv = *(const float4*)(in + i);
    v4us o;
    o[0] = f2bf(vv.x); o[1] = f2bf(vv.y); o[2] = f2bf(vv.z); o[3] = f2bf(vv.w);
    *(v4us*)(out + i) = o;
  }
}

// -------------------------------------------------- A-fragment load (16x32, MxK)
template <int ABF16>
__device__ __forceinline__ v16us load_afrag(const void* Aptr, size_t rowoff,
                                            int k0, int kb) {
  if (ABF16) {
    const unsigned short* p = (const unsigned short*)Aptr + rowoff + k0 + kb;
    v8us alo = *(const v8us*)(p);
    v8us ahi = *(const v8us*)(p + 8);
    return __builtin_shufflevector(alo, ahi, 0,1,2,3,4,5,6,7,8,9,10,11,12,13,14,15);
  } else {
    const float* p = (const float*)Aptr + rowoff + k0 + kb;
    v8f flo = *(const v8f*)(p);
    v8f fhi = *(const v8f*)(p + 8);
    v16us a;
    #pragma unroll
    for (int i = 0; i < 8; ++i) { a[i] = f2bf(flo[i]); a[i + 8] = f2bf(fhi[i]); }
    return a;
  }
}

// ---------------------------------------------------------------- GEMM kernel
// C[m,n] = sum_k A[m,k] * W[n,k] + bias[n]    (A: M_ x D_, W: D_ x D_ bf16)
// MODE 0: bf16 out [B,H,S,DH]; MODE 1: bf16 out [B,H,DH,S]; MODE 2: f32 out [M,D]
// Per-wave tile: 32(M) x 128(N), 16 accumulators. The 128x32 W chunk (8 KB) is
// shared by all 8 waves: staged into LDS with async-to-LDS DMA, double-buffered.
template <int ABF16, int MODE>
__global__ __launch_bounds__(256) void gemm_wmma(
    const void* __restrict__ Aptr,
    const unsigned short* __restrict__ Wb,
    const float* __restrict__ bias,
    void* __restrict__ Out) {
  __shared__ __align__(32) unsigned short wlds[2][128 * 32];  // 2 x 8 KB

  const int lane = threadIdx.x & 31;
  const int wave = threadIdx.x >> 5;
  const int hlf  = lane >> 4;       // 0/1 half of wave
  const int lo   = lane & 15;
  const int kb   = hlf * 16;

  const int m0 = (blockIdx.y * 8 + wave) * 32;
  const int n0 = blockIdx.x * 128;

  const size_t rowA0 = (size_t)(m0 + lo) * D_;
  const size_t rowA1 = (size_t)(m0 + 16 + lo) * D_;
  const int abytes = ABF16 ? 2 : 4;

  // staging map: thread t copies 32 B of row (t>>1), segment (t&1)*2 .. +1
  const int srow = threadIdx.x >> 1;
  const int sseg = (threadIdx.x & 1) * 2;
  const unsigned short* gsb = Wb + (size_t)(n0 + srow) * D_ + sseg * 8;
  const unsigned ldsA =
      (unsigned)(uintptr_t)(lds_us*)&wlds[0][srow * 32 + sseg * 8];
  const unsigned ldsB =
      (unsigned)(uintptr_t)(lds_us*)&wlds[1][srow * 32 + sseg * 8];

  v8f acc0[8] = {};
  v8f acc1[8] = {};

  // prologue: stage chunk 0 into buffer 0
  async_ld_b128(ldsA,      gsb);
  async_ld_b128(ldsA + 16, gsb + 8);

  for (int c = 0; c < D_ / 32; ++c) {
    const int k0 = c * 32;
    wait_async0();        // our staging loads for buffer c&1 done
    __syncthreads();      // everyone's staging done / prev reads done

    if (c + 1 < D_ / 32) {  // stage next chunk into other buffer
      const unsigned l = ((c + 1) & 1) ? ldsB : ldsA;
      async_ld_b128(l,      gsb + (c + 1) * 32);
      async_ld_b128(l + 16, gsb + (c + 1) * 32 + 8);
    }

    __builtin_prefetch((const char*)Aptr + (rowA0 + k0 + 256) * abytes, 0, 3);
    __builtin_prefetch((const char*)Aptr + (rowA1 + k0 + 256) * abytes, 0, 3);
    v16us a0 = load_afrag<ABF16>(Aptr, rowA0, k0, kb);
    v16us a1 = load_afrag<ABF16>(Aptr, rowA1, k0, kb);

    const unsigned short* wc = &wlds[c & 1][0];
    #pragma unroll
    for (int nn = 0; nn < 8; ++nn) {
      v16us b = *(const v16us*)(wc + (nn * 16 + lo) * 32 + hlf * 16);
      acc0[nn] = wmma_bf16(a0, b, acc0[nn]);
      acc1[nn] = wmma_bf16(a1, b, acc1[nn]);
    }
  }

  #pragma unroll
  for (int nn = 0; nn < 8; ++nn) {
    const int n = n0 + nn * 16 + lo;
    const float bs = bias[n];
    #pragma unroll
    for (int half2 = 0; half2 < 2; ++half2) {
      const v8f* acc = half2 ? acc1 : acc0;
      #pragma unroll
      for (int r = 0; r < 8; ++r) {
        const int m = m0 + half2 * 16 + hlf * 8 + r;
        const float val = acc[nn][r] + bs;
        if (MODE == 0) {
          int bb = m >> 11, s = m & (S_ - 1);
          int h = n >> 7,  dh = n & (DH_ - 1);
          ((unsigned short*)Out)[((size_t)(bb * H_ + h) * S_ + s) * DH_ + dh] = f2bf(val);
        } else if (MODE == 1) {
          int bb = m >> 11, s = m & (S_ - 1);
          int h = n >> 7,  dh = n & (DH_ - 1);
          ((unsigned short*)Out)[((size_t)(bb * H_ + h) * DH_ + dh) * S_ + s] = f2bf(val);
        } else {
          ((float*)Out)[(size_t)m * D_ + n] = val;
        }
      }
    }
  }
}

// ------------------------------------------------------- flash attention kernel
// Qp,Kp: [BH][S][DH] bf16 ; Vt: [BH][DH][S] bf16 ; Aout: [B][S][D] bf16
__global__ __launch_bounds__(256) void flash_attn(
    const unsigned short* __restrict__ Qp,
    const unsigned short* __restrict__ Kp,
    const unsigned short* __restrict__ Vt,
    unsigned short* __restrict__ Aout) {
  __shared__ __align__(32) unsigned short plds[8][16 * 32];

  const int lane = threadIdx.x & 31;
  const int wave = threadIdx.x >> 5;
  const int hlf  = lane >> 4;
  const int lo   = lane & 15;
  const int kb   = hlf * 16;

  const int t  = blockIdx.x * 8 + wave;
  const int bh = t >> 7;           // 32 (b,h) pairs
  const int qt = t & 127;          // 128 q-tiles of 16 rows
  const int q0 = qt * 16;
  const int bb = bh >> 4, h = bh & 15;

  const unsigned short* qbase = Qp + ((size_t)bh * S_ + q0 + lo) * DH_;
  v16us qf[4];
  #pragma unroll
  for (int c = 0; c < 4; ++c) {
    v8us alo = *(const v8us*)(qbase + c * 32 + kb);
    v8us ahi = *(const v8us*)(qbase + c * 32 + kb + 8);
    qf[c] = __builtin_shufflevector(alo, ahi, 0,1,2,3,4,5,6,7,8,9,10,11,12,13,14,15);
  }

  v8f o[8] = {};
  float mrow[8], lrow[8];
  #pragma unroll
  for (int r = 0; r < 8; ++r) { mrow[r] = -INFINITY; lrow[r] = 0.0f; }

  const float sc = 0.08838834764831844f;  // 1/sqrt(128)
  const int jtmax = (q0 + 15) >> 5;       // causal: kv tiles of 32

  for (int jt = 0; jt <= jtmax; ++jt) {
    const int j0 = jt * 32;
    v8f s0 = {}, s1 = {};
    const unsigned short* kbase = Kp + ((size_t)bh * S_ + j0) * DH_;
    #pragma unroll
    for (int c = 0; c < 4; ++c) {
      v16us b0 = *(const v16us*)(kbase + (size_t)lo * DH_ + c * 32 + hlf * 16);
      v16us b1 = *(const v16us*)(kbase + (size_t)(16 + lo) * DH_ + c * 32 + hlf * 16);
      s0 = wmma_bf16(qf[c], b0, s0);
      s1 = wmma_bf16(qf[c], b1, s1);
    }

    float corr[8];
    #pragma unroll
    for (int r = 0; r < 8; ++r) {
      const int row = q0 + hlf * 8 + r;
      float x0 = s0[r] * sc;
      float x1 = s1[r] * sc;
      if (j0 + lo > row)      x0 = -INFINITY;   // causal mask
      if (j0 + 16 + lo > row) x1 = -INFINITY;
      float tm = fmaxf(x0, x1);
      tm = fmaxf(tm, __shfl_xor(tm, 1, 32));
      tm = fmaxf(tm, __shfl_xor(tm, 2, 32));
      tm = fmaxf(tm, __shfl_xor(tm, 4, 32));
      tm = fmaxf(tm, __shfl_xor(tm, 8, 32));
      const float mn = fmaxf(mrow[r], tm);
      const float cr = __expf(mrow[r] - mn);
      const float p0 = __expf(x0 - mn);
      const float p1 = __expf(x1 - mn);
      float ts = p0 + p1;
      ts += __shfl_xor(ts, 1, 32);
      ts += __shfl_xor(ts, 2, 32);
      ts += __shfl_xor(ts, 4, 32);
      ts += __shfl_xor(ts, 8, 32);
      lrow[r] = lrow[r] * cr + ts;
      mrow[r] = mn;
      corr[r] = cr;
      const int prow = hlf * 8 + r;             // C-layout -> LDS row-major
      plds[wave][prow * 32 + lo]      = f2bf(p0);
      plds[wave][prow * 32 + 16 + lo] = f2bf(p1);
    }

    #pragma unroll
    for (int nn = 0; nn < 8; ++nn)
      #pragma unroll
      for (int r = 0; r < 8; ++r)
        o[nn][r] *= corr[r];

    asm volatile("s_wait_dscnt 0" ::: "memory");  // same-wave LDS RAW

    v8us plo = *(const v8us*)&plds[wave][lo * 32 + kb];
    v8us phi = *(const v8us*)&plds[wave][lo * 32 + kb + 8];
    v16us pa = __builtin_shufflevector(plo, phi, 0,1,2,3,4,5,6,7,8,9,10,11,12,13,14,15);

    const unsigned short* vb0 = Vt + (size_t)bh * DH_ * S_ + j0 + hlf * 16;
    #pragma unroll
    for (int nn = 0; nn < 8; ++nn) {
      const int dh = nn * 16 + lo;
      v16us vb = *(const v16us*)(vb0 + (size_t)dh * S_);
      o[nn] = wmma_bf16(pa, vb, o[nn]);
    }
  }

  float rinv[8];
  #pragma unroll
  for (int r = 0; r < 8; ++r) rinv[r] = __frcp_rn(lrow[r]);

  #pragma unroll
  for (int nn = 0; nn < 8; ++nn) {
    #pragma unroll
    for (int r = 0; r < 8; ++r) {
      const int row = q0 + hlf * 8 + r;
      Aout[((size_t)(bb * S_ + row)) * D_ + h * DH_ + nn * 16 + lo] =
          f2bf(o[nn][r] * rinv[r]);
    }
  }
}

// ---------------------------------------------------------------- launcher
extern "C" void kernel_launch(void* const* d_in, const int* in_sizes, int n_in,
                              void* d_out, int out_size, void* d_ws, size_t ws_size,
                              hipStream_t stream) {
  const float* q  = (const float*)d_in[0];
  const float* k  = (const float*)d_in[1];
  const float* v  = (const float*)d_in[2];
  const float* Wq = (const float*)d_in[3];
  const float* bq = (const float*)d_in[4];
  const float* Wk = (const float*)d_in[5];
  const float* bk = (const float*)d_in[6];
  const float* Wv = (const float*)d_in[7];
  const float* bv = (const float*)d_in[8];
  const float* Wo = (const float*)d_in[9];
  const float* bo = (const float*)d_in[10];
  // d_in[11] = causal_mask (always 1 in the reference)

  char* ws = (char*)d_ws;
  const size_t SZ_PROJ = (size_t)BH_ * S_ * DH_ * 2;            // 16 MiB each
  unsigned short* Qp  = (unsigned short*)(ws);
  unsigned short* Kp  = (unsigned short*)(ws + SZ_PROJ);
  unsigned short* Vt  = (unsigned short*)(ws + 2 * SZ_PROJ);
  unsigned short* Ao  = (unsigned short*)(ws + 3 * SZ_PROJ);
  unsigned short* Wqb = (unsigned short*)(ws + 4 * SZ_PROJ);
  unsigned short* Wkb = Wqb + (size_t)D_ * D_;
  unsigned short* Wvb = Wkb + (size_t)D_ * D_;
  unsigned short* Wob = Wvb + (size_t)D_ * D_;

  const int ncast = D_ * D_;
  dim3 cgrid(ncast / (256 * 4));
  cast_f32_bf16<<<cgrid, 256, 0, stream>>>(Wq, Wqb, ncast);
  cast_f32_bf16<<<cgrid, 256, 0, stream>>>(Wk, Wkb, ncast);
  cast_f32_bf16<<<cgrid, 256, 0, stream>>>(Wv, Wvb, ncast);
  cast_f32_bf16<<<cgrid, 256, 0, stream>>>(Wo, Wob, ncast);

  dim3 ggrid(D_ / 128, M_ / (32 * 8));   // (16, 16), 8 waves/block, 32x128/wave
  gemm_wmma<0, 0><<<ggrid, 256, 0, stream>>>(q, Wqb, bq, Qp);
  gemm_wmma<0, 0><<<ggrid, 256, 0, stream>>>(k, Wkb, bk, Kp);
  gemm_wmma<0, 1><<<ggrid, 256, 0, stream>>>(v, Wvb, bv, Vt);

  flash_attn<<<dim3(BH_ * (S_ / 16) / 8), 256, 0, stream>>>(Qp, Kp, Vt, Ao);

  gemm_wmma<1, 2><<<ggrid, 256, 0, stream>>>(Ao, Wob, bo, d_out);
}
